// DifferentialCrossAttention_12610023981459
// MI455X (gfx1250) — matrix operand order repaired and checked by reading, not compiled, not run on hardware
//
#include <hip/hip_runtime.h>
#include <stdint.h>

// ---------------------------------------------------------------------------
// Differential cross attention for MI455X (gfx1250), bf16 WMMA path.
// B=4, Nq=Nkv=1024, DIM=512, H=8, HD=64, MAX_DIST=128, LAMBDA_INIT=0.8
// ---------------------------------------------------------------------------

typedef __bf16 bf16x16 __attribute__((ext_vector_type(16)));
typedef float  f32x8   __attribute__((ext_vector_type(8)));

struct U128 { uint32_t x, y, z, w; };
struct U256 { U128 lo, hi; };

__device__ inline uint16_t f2bf(float f) {
  uint32_t u = __builtin_bit_cast(uint32_t, f);
  u += 0x7FFFu + ((u >> 16) & 1u);          // round-to-nearest-even
  return (uint16_t)(u >> 16);
}

// A-fragment (16x32 bf16): lane holds 16 contiguous K values at p
// (lane-half selects K 0..15 or 16..31).
__device__ inline bf16x16 frag_ld_split(const uint16_t* p) {
  U256 t;
  t.lo = *(const U128*)(p);
  t.hi = *(const U128*)(p + 8);
  return __builtin_bit_cast(bf16x16, t);
}
// B-fragment (32x16 bf16): lane = column, 16 contiguous K values.
__device__ inline bf16x16 frag_ld_contig(const uint16_t* p) {
  U256 t;
  t.lo = *(const U128*)(p);
  t.hi = *(const U128*)(p + 8);
  return __builtin_bit_cast(bf16x16, t);
}

__device__ inline f32x8 vzero8() {
  f32x8 v = {0.f, 0.f, 0.f, 0.f, 0.f, 0.f, 0.f, 0.f};
  return v;
}

__device__ inline f32x8 wmma_bf16(bf16x16 a, bf16x16 b, f32x8 c) {
  return __builtin_amdgcn_wmma_f32_16x16x32_bf16(
      /*neg_a=*/false, a, /*neg_b=*/false, b,
      /*c_mod=*/(short)0, c, /*reuse_a=*/false, /*reuse_b=*/false);
}

__device__ inline float rmax16(float v) {
  for (int off = 1; off < 16; off <<= 1) v = fmaxf(v, __shfl_xor(v, off, 32));
  return v;
}
__device__ inline float rsum16(float v) {
  for (int off = 1; off < 16; off <<= 1) v += __shfl_xor(v, off, 32);
  return v;
}
__device__ inline int iclamp(int x, int lo, int hi) {
  return x < lo ? lo : (x > hi ? hi : x);
}

// ---------------------------------------------------------------------------
// fp32 -> bf16 conversion
// ---------------------------------------------------------------------------
__global__ void cvt_bf16_kernel(const float* __restrict__ in,
                                uint16_t* __restrict__ out, int n) {
  int i = blockIdx.x * blockDim.x + threadIdx.x;
  if (i < n) out[i] = f2bf(in[i]);
}

// ---------------------------------------------------------------------------
// lambda per head pair: exp(sum lq1*lk1) - exp(sum lq2*lk2) + 0.8
// ---------------------------------------------------------------------------
__global__ void lam_kernel(const float* __restrict__ lq1, const float* __restrict__ lk1,
                           const float* __restrict__ lq2, const float* __restrict__ lk2,
                           float* __restrict__ lam) {
  __shared__ float s1[64], s2[64];
  int p = blockIdx.x, t = threadIdx.x;
  s1[t] = lq1[p * 64 + t] * lk1[p * 64 + t];
  s2[t] = lq2[p * 64 + t] * lk2[p * 64 + t];
  __syncthreads();
  for (int off = 32; off > 0; off >>= 1) {
    if (t < off) { s1[t] += s1[t + off]; s2[t] += s2[t + off]; }
    __syncthreads();
  }
  if (t == 0) lam[p] = __expf(s1[0]) - __expf(s2[0]) + 0.8f;
}

// ---------------------------------------------------------------------------
// GEMM: out[m,n] = sum_k A[m,k] * W[n,k] + bias[n]     (A,W bf16, acc f32)
// MODE 0: bf16 out, row-major (m*N + n)
// MODE 1: bf16 out, per-head transposed V layout: Vt[(b*8+h)*64+d][key]
// MODE 2: f32 out, row-major
// One wave -> one 16x16 tile; block = 4 waves = 4 consecutive n-tiles.
// ---------------------------------------------------------------------------
template <int MODE>
__global__ __launch_bounds__(128) void gemm_bf16_kernel(
    const uint16_t* __restrict__ A, const uint16_t* __restrict__ W,
    const float* __restrict__ bias, void* __restrict__ outp,
    int M, int N, int K) {
  const int lane = threadIdx.x & 31;
  const int wv   = threadIdx.x >> 5;
  const int n0   = (blockIdx.x * 4 + wv) * 16;
  const int m0   = blockIdx.y * 16;
  const int half = lane >> 4, col = lane & 15;

  f32x8 acc = vzero8();
  const uint16_t* ar = A + (size_t)(m0 + col) * K;   // A fragment row = col
  const uint16_t* wr = W + (size_t)(n0 + col) * K;   // B fragment col = col
  for (int k0 = 0; k0 < K; k0 += 32) {
    bf16x16 af  = frag_ld_split(ar + k0 + half * 16);
    bf16x16 bfr = frag_ld_contig(wr + k0 + half * 16);
    acc = wmma_bf16(af, bfr, acc);
  }

  for (int r = 0; r < 8; ++r) {
    int m = m0 + r + 8 * half, n = n0 + col;
    float v = acc[r] + bias[n];
    if (MODE == 0) {
      ((uint16_t*)outp)[(size_t)m * N + n] = f2bf(v);
    } else if (MODE == 1) {
      int bb = m >> 10, key = m & 1023, h = n >> 6, d = n & 63;
      ((uint16_t*)outp)[(((size_t)(bb * 8 + h) * 64 + d) << 10) + key] = f2bf(v);
    } else {
      ((float*)outp)[(size_t)m * N + n] = v;
    }
  }
}

// ---------------------------------------------------------------------------
// Flash differential attention. One wave: 16 queries x head pair (h, h+4).
// ---------------------------------------------------------------------------
__global__ __launch_bounds__(128) void attn_kernel(
    const uint16_t* __restrict__ Qb, const uint16_t* __restrict__ Kb,
    const uint16_t* __restrict__ Vt, const int* __restrict__ cq,
    const int* __restrict__ ck, const float* __restrict__ alpha_map,
    const float* __restrict__ rpe, const float* __restrict__ lam_ws,
    uint16_t* __restrict__ attO) {
  __shared__ __align__(16) uint16_t pstage[4][16 * 32];  // per-wave P transpose

  const int lane = threadIdx.x & 31;
  const int w    = threadIdx.x >> 5;
  const int g    = blockIdx.x * 4 + w;          // 0..1023
  const int qt   = g & 63;
  const int pair = (g >> 6) & 3;
  const int b    = g >> 8;
  const int q0   = qt * 16;
  const int half = lane >> 4;
  const int col  = lane & 15;
  const int h0 = pair, h1 = pair + 4;
  const float scale = 0.125f;                   // 64^-0.5

  // Q fragments for both heads (A layout, K = head dim, split in 2x32)
  bf16x16 aq0[2], aq1[2];
  {
    const uint16_t* qrow = Qb + (size_t)(b * 1024 + q0 + col) * 512;
    for (int c = 0; c < 2; ++c) {
      aq0[c] = frag_ld_split(qrow + h0 * 64 + c * 32 + half * 16);
      aq1[c] = frag_ld_split(qrow + h1 * 64 + c * 32 + half * 16);
    }
  }

  int qcx[8], qcy[8];
  float alp[8];
  for (int r = 0; r < 8; ++r) {
    int q = q0 + r + 8 * half;                  // rows this lane owns (C layout)
    qcx[r] = cq[(b * 1024 + q) * 2 + 0];
    qcy[r] = cq[(b * 1024 + q) * 2 + 1];
    alp[r] = alpha_map[b * 1024 + q];
  }

  float m0[8], l0[8], m1[8], l1[8];
  for (int r = 0; r < 8; ++r) { m0[r] = m1[r] = -1e30f; l0[r] = l1[r] = 0.f; }
  f32x8 O0[4], Dd[4], O4[4];
  for (int dt = 0; dt < 4; ++dt) { O0[dt] = vzero8(); Dd[dt] = vzero8(); O4[dt] = vzero8(); }

  for (int kb = 0; kb < 1024; kb += 32) {
    // ---- scores: S = Q @ K^T (two 16-key tiles, both heads) ----
    f32x8 s0a = vzero8(), s0b = vzero8(), s1a = vzero8(), s1b = vzero8();
    const uint16_t* kr0 = Kb + (size_t)(b * 1024 + kb + col) * 512;
    const uint16_t* kr1 = Kb + (size_t)(b * 1024 + kb + 16 + col) * 512;
    for (int c = 0; c < 2; ++c) {
      bf16x16 b00 = frag_ld_contig(kr0 + h0 * 64 + c * 32 + half * 16);
      bf16x16 b01 = frag_ld_contig(kr1 + h0 * 64 + c * 32 + half * 16);
      bf16x16 b10 = frag_ld_contig(kr0 + h1 * 64 + c * 32 + half * 16);
      bf16x16 b11 = frag_ld_contig(kr1 + h1 * 64 + c * 32 + half * 16);
      s0a = wmma_bf16(aq0[c], b00, s0a);
      s0b = wmma_bf16(aq0[c], b01, s0b);
      s1a = wmma_bf16(aq1[c], b10, s1a);
      s1b = wmma_bf16(aq1[c], b11, s1b);
    }

    // ---- relative-position bias gather ----
    int kx0 = ck[(b * 1024 + kb + col) * 2 + 0];
    int ky0 = ck[(b * 1024 + kb + col) * 2 + 1];
    int kx1 = ck[(b * 1024 + kb + 16 + col) * 2 + 0];
    int ky1 = ck[(b * 1024 + kb + 16 + col) * 2 + 1];
    float f0a[8], f0b[8], f1a[8], f1b[8];
    for (int r = 0; r < 8; ++r) {
      int i0 = iclamp(qcx[r] - kx0 + 128, 0, 256) * 257 + iclamp(qcy[r] - ky0 + 128, 0, 256);
      int i1 = iclamp(qcx[r] - kx1 + 128, 0, 256) * 257 + iclamp(qcy[r] - ky1 + 128, 0, 256);
      f0a[r] = s0a[r] * scale + rpe[i0 * 8 + h0];
      f0b[r] = s0b[r] * scale + rpe[i1 * 8 + h0];
      f1a[r] = s1a[r] * scale + rpe[i0 * 8 + h1];
      f1b[r] = s1b[r] * scale + rpe[i1 * 8 + h1];
    }

    // ---- online softmax head h0, stage P into LDS (C->A transpose) ----
    float corr0[8], corr1[8];
    for (int r = 0; r < 8; ++r) {
      float mx = rmax16(fmaxf(f0a[r], f0b[r]));
      float nm = fmaxf(m0[r], mx);
      corr0[r] = __expf(m0[r] - nm);
      m0[r] = nm;
      float pa = __expf(f0a[r] - nm);
      float pb = __expf(f0b[r] - nm);
      l0[r] = l0[r] * corr0[r] + rsum16(pa + pb);
      int row = r + 8 * half;
      pstage[w][row * 32 + col]      = f2bf(pa);
      pstage[w][row * 32 + col + 16] = f2bf(pb);
    }
    asm volatile("s_wait_dscnt 0" ::: "memory");
    bf16x16 pa0 = frag_ld_split(&pstage[w][col * 32 + half * 16]);

    // ---- online softmax head h1 ----
    for (int r = 0; r < 8; ++r) {
      float mx = rmax16(fmaxf(f1a[r], f1b[r]));
      float nm = fmaxf(m1[r], mx);
      corr1[r] = __expf(m1[r] - nm);
      m1[r] = nm;
      float pa = __expf(f1a[r] - nm);
      float pb = __expf(f1b[r] - nm);
      l1[r] = l1[r] * corr1[r] + rsum16(pa + pb);
      int row = r + 8 * half;
      pstage[w][row * 32 + col]      = f2bf(pa);
      pstage[w][row * 32 + col + 16] = f2bf(pb);
    }
    asm volatile("s_wait_dscnt 0" ::: "memory");
    bf16x16 pa1 = frag_ld_split(&pstage[w][col * 32 + half * 16]);

    // ---- P @ V accumulation (V stored transposed: contiguous keys) ----
    for (int dt = 0; dt < 4; ++dt) {
      for (int r = 0; r < 8; ++r) {
        O0[dt][r] *= corr0[r];
        Dd[dt][r] *= corr1[r];
        O4[dt][r] *= corr1[r];
      }
      const uint16_t* v0 = Vt + ((size_t)((b * 8 + h0) * 64 + dt * 16 + col) << 10) + kb + half * 16;
      const uint16_t* v1 = Vt + ((size_t)((b * 8 + h1) * 64 + dt * 16 + col) << 10) + kb + half * 16;
      bf16x16 vb0 = frag_ld_contig(v0);
      bf16x16 vb1 = frag_ld_contig(v1);
      O0[dt] = wmma_bf16(pa0, vb0, O0[dt]);
      Dd[dt] = wmma_bf16(pa1, vb0, Dd[dt]);
      O4[dt] = wmma_bf16(pa1, vb1, O4[dt]);
    }
  }

  // ---- differential combine + store bf16 attention output ----
  float lam = lam_ws[pair];
  for (int dt = 0; dt < 4; ++dt) {
    for (int r = 0; r < 8; ++r) {
      int q = q0 + r + 8 * half;
      int d = dt * 16 + col;
      float i0 = 1.f / l0[r], i1 = 1.f / l1[r];
      float oh = O0[dt][r] * i0;
      float dd = Dd[dt][r] * i1;
      float o4 = O4[dt][r] * i1;
      float oo = (1.f + alp[r]) * oh - alp[r] * lam * dd;
      size_t base = (size_t)(b * 1024 + q) * 512;
      attO[base + h0 * 64 + d] = f2bf(oo);
      attO[base + h1 * 64 + d] = f2bf(o4);
    }
  }
}

// ---------------------------------------------------------------------------
// Host-side launch
// ---------------------------------------------------------------------------
extern "C" void kernel_launch(void* const* d_in, const int* in_sizes, int n_in,
                              void* d_out, int out_size, void* d_ws, size_t ws_size,
                              hipStream_t stream) {
  (void)in_sizes; (void)n_in; (void)out_size; (void)ws_size;
  const float* x_q   = (const float*)d_in[0];
  const float* x_kv  = (const float*)d_in[1];
  const int*   cq    = (const int*)d_in[2];
  const int*   ck    = (const int*)d_in[3];
  const float* alpha = (const float*)d_in[4];
  const float* Wq    = (const float*)d_in[5];
  const float* bq    = (const float*)d_in[6];
  const float* Wk    = (const float*)d_in[7];
  const float* bk    = (const float*)d_in[8];
  const float* Wv    = (const float*)d_in[9];
  const float* bv    = (const float*)d_in[10];
  const float* lq1   = (const float*)d_in[11];
  const float* lk1   = (const float*)d_in[12];
  const float* lq2   = (const float*)d_in[13];
  const float* lk2   = (const float*)d_in[14];
  const float* rpe   = (const float*)d_in[15];
  const float* Wp    = (const float*)d_in[16];
  const float* bp    = (const float*)d_in[17];

  const int M = 4096, N = 512, K = 512;
  const size_t SZ_X = (size_t)M * N * 2;   // 4 MiB (bf16 4096x512)
  const size_t SZ_W = (size_t)N * K * 2;   // 512 KiB

  char* ws = (char*)d_ws;
  uint16_t* xq_b  = (uint16_t*)(ws);
  uint16_t* xkv_b = (uint16_t*)(ws + SZ_X);
  uint16_t* wq_b  = (uint16_t*)(ws + 2 * SZ_X);
  uint16_t* wk_b  = (uint16_t*)(ws + 2 * SZ_X + SZ_W);
  uint16_t* wv_b  = (uint16_t*)(ws + 2 * SZ_X + 2 * SZ_W);
  uint16_t* wp_b  = (uint16_t*)(ws + 2 * SZ_X + 3 * SZ_W);
  uint16_t* q_b   = (uint16_t*)(ws + 2 * SZ_X + 4 * SZ_W);
  uint16_t* k_b   = (uint16_t*)(ws + 3 * SZ_X + 4 * SZ_W);
  uint16_t* vt_b  = (uint16_t*)(ws + 4 * SZ_X + 4 * SZ_W);
  uint16_t* ao_b  = (uint16_t*)(ws + 5 * SZ_X + 4 * SZ_W);
  float*    lam_w = (float*)   (ws + 6 * SZ_X + 4 * SZ_W);

  // fp32 -> bf16 conversions
  {
    int n = M * K;  // 2097152
    cvt_bf16_kernel<<<(n + 255) / 256, 256, 0, stream>>>(x_q,  xq_b,  n);
    cvt_bf16_kernel<<<(n + 255) / 256, 256, 0, stream>>>(x_kv, xkv_b, n);
    int nw = N * K; // 262144
    cvt_bf16_kernel<<<(nw + 255) / 256, 256, 0, stream>>>(Wq, wq_b, nw);
    cvt_bf16_kernel<<<(nw + 255) / 256, 256, 0, stream>>>(Wk, wk_b, nw);
    cvt_bf16_kernel<<<(nw + 255) / 256, 256, 0, stream>>>(Wv, wv_b, nw);
    cvt_bf16_kernel<<<(nw + 255) / 256, 256, 0, stream>>>(Wp, wp_b, nw);
  }

  // lambda per head pair
  lam_kernel<<<4, 64, 0, stream>>>(lq1, lk1, lq2, lk2, lam_w);

  // Q, K, V projections (WMMA GEMMs)
  dim3 ggrid(N / 64, M / 16);
  gemm_bf16_kernel<0><<<ggrid, 128, 0, stream>>>(xq_b,  wq_b, bq, q_b,  M, N, K);
  gemm_bf16_kernel<0><<<ggrid, 128, 0, stream>>>(xkv_b, wk_b, bk, k_b,  M, N, K);
  gemm_bf16_kernel<1><<<ggrid, 128, 0, stream>>>(xkv_b, wv_b, bv, vt_b, M, N, K);

  // flash differential attention: 1024 wave-tiles, 4 waves/block
  attn_kernel<<<256, 128, 0, stream>>>(q_b, k_b, vt_b, cq, ck, alpha, rpe,
                                       lam_w, ao_b);

  // output projection -> fp32 d_out
  gemm_bf16_kernel<2><<<ggrid, 128, 0, stream>>>(ao_b, wp_b, bp, d_out, M, N, K);
}
